// VoxelSEResBackBone8x_85383949844629
// MI455X (gfx1250) — matrix operand order repaired and checked by reading, not compiled, not run on hardware
//
#include <hip/hip_runtime.h>
#include <hip/hip_bf16.h>

typedef __attribute__((ext_vector_type(16))) _Float16 v16h;
typedef __attribute__((ext_vector_type(8)))  _Float16 v8h;
typedef __attribute__((ext_vector_type(8)))  float    v8f;

#define BN_EPS 1e-3f

// ---------------------------------------------------------------------------
// Implicit-GEMM 3D convolution with fused bias+BN+ReLU+mask epilogue.
// One wave computes 32 output voxels (one x-run of an output row) x 16 output
// channels, using 2x v_wmma_f32_16x16x32_f16 per K-chunk (shared B fragment).
// Window shape is a template parameter so all k-decomposition divisions use
// compile-time divisors; grid mapping removes per-element voxel division.
// ---------------------------------------------------------------------------
struct ConvParams {
    const float* in;     // [B,Cin,Di,Hi,Wi]
    float*       out;    // [B,Cout,Do,Ho,Wo]
    const float* w;      // [Cout,Cin,kd,kh,kw]
    const float* bias;   // [Cout] or null
    const float* gamma;  // BN params [Cout]
    const float* beta;
    const float* mean;
    const float* var;
    const float* mask;   // [B, Do*Ho*Wo] or null
    int Cin, Cout;
    int Di, Hi, Wi;
    int Do, Ho, Wo;
    int sd, sh, sw;
    int pd, ph, pw;
    int relu;
    int nXTiles;         // ceil(Wo/32)
};

template <int KD, int KH, int KW>
__global__ __launch_bounds__(32) void conv3d_wmma_kernel(ConvParams p) {
    constexpr int KTAPS = KD * KH * KW;
    constexpr int KHW   = KH * KW;

    const int lane = threadIdx.x;                 // 0..31, full wave
    const int bx   = blockIdx.x;
    const int coTile = bx / p.nXTiles;            // uniform scalar div (per block)
    const int xTile  = bx - coTile * p.nXTiles;
    const int by   = blockIdx.y;                  // z*Ho + y
    const int z    = by / p.Ho;                   // uniform scalar div (per block)
    const int y    = by - z * p.Ho;
    const int b    = blockIdx.z;

    const int x0   = xTile * 32;
    const int co0  = coTile * 16;
    const int Ktot = p.Cin * KTAPS;

    const long HiWi  = (long)p.Hi * p.Wi;
    const long CinKT = (long)p.Cin * KTAPS;

    __shared__ __align__(16) _Float16 Asm[32 * 32];  // [m 0..31][k 0..31]
    __shared__ __align__(16) _Float16 Bsm[16 * 32];  // [n 0..15][k 0..31]

    v8f acc0 = {0.f, 0.f, 0.f, 0.f, 0.f, 0.f, 0.f, 0.f};
    v8f acc1 = {0.f, 0.f, 0.f, 0.f, 0.f, 0.f, 0.f, 0.f};

    for (int k0 = 0; k0 < Ktot; k0 += 32) {
        // Per-chunk, per-lane k decomposition (compile-time divisors).
        const int k   = k0 + lane;
        const int ci  = k / KTAPS;
        const int t   = k - ci * KTAPS;
        const int kd_ = t / KHW;
        const int t2  = t - kd_ * KHW;
        const int kh_ = t2 / KW;
        const int kw_ = t2 - kh_ * KW;
        const int zi  = z * p.sd - p.pd + kd_;
        const int yi  = y * p.sh - p.ph + kh_;
        const bool vk = (k < Ktot) && (zi >= 0) && (zi < p.Di) &&
                        (yi >= 0) && (yi < p.Hi);
        const float* src =
            p.in + (((long)b * p.Cin + ci) * p.Di + zi) * HiWi + (long)yi * p.Wi;

        // ---- stage A: lane owns k-column `lane`, walks 32 voxels along x ----
        int xi = x0 * p.sw - p.pw + kw_;
        #pragma unroll
        for (int it = 0; it < 32; ++it) {         // uniform trip count
            float v = 0.f;
            if (vk && xi >= 0 && xi < p.Wi) v = src[xi];
            Asm[it * 32 + lane] = (_Float16)v;
            xi += p.sw;
        }
        // ---- stage B: lane owns k-column `lane`, walks 16 output channels ----
        long wi = (long)co0 * CinKT + (long)ci * KTAPS + t;
        #pragma unroll
        for (int it = 0; it < 16; ++it) {         // uniform trip count
            float v = 0.f;
            if (k < Ktot) v = p.w[wi];
            Bsm[it * 32 + lane] = (_Float16)v;
            wi += CinKT;
        }
        __syncthreads();

        // ---- fragments (contiguous 16B runs -> ds_load_b128) ----
        // A 16x32: lanes<16 -> M=lane, K in {0..7,16..23}; lanes>=16 -> +8
        // B 32x16: lanes<16 -> N=lane, K in {0..7,16..23}; lanes>=16 -> N=lane-16, +8
        const int base = (lane >= 16) ? 8 : 0;
        const int mrow = lane & 15;
        const int kofs = (lane >= 16) ? 16 : 0;

        v8h a0lo = *(const v8h*)(Asm + mrow * 32 + base);
        v8h a0hi = *(const v8h*)(Asm + mrow * 32 + 16 + base);
        v8h a1lo = *(const v8h*)(Asm + (16 + mrow) * 32 + base);
        v8h a1hi = *(const v8h*)(Asm + (16 + mrow) * 32 + 16 + base);
        v8h blo = *(const v8h*)(Bsm + mrow * 32 + base);
        v8h bhi = *(const v8h*)(Bsm + mrow * 32 + 16 + base);
        (void)kofs;

        v16h a0 = {}, a1 = {}, bfrag = {};
        #pragma unroll
        for (int h = 0; h < 8; ++h) {
            a0[h] = a0lo[h]; a0[8 + h] = a0hi[h];
            a1[h] = a1lo[h]; a1[8 + h] = a1hi[h];
            bfrag[h] = blo[h]; bfrag[8 + h] = bhi[h];
        }

        acc0 = __builtin_amdgcn_wmma_f32_16x16x32_f16(
            false, a0, false, bfrag, (short)0, acc0, false, false);
        acc1 = __builtin_amdgcn_wmma_f32_16x16x32_f16(
            false, a1, false, bfrag, (short)0, acc1, false, false);
        __syncthreads();
    }

    // ---- epilogue: (x + bias)*bn_scale + bn_shift, ReLU, mask ----
    // C layout: VGPR r -> M = r (lanes<16) / 8+r (lanes>=16), N = lane%16
    const int  n       = lane & 15;
    const int  co      = co0 + n;                  // Cout is a multiple of 16
    const int  mb      = (lane >= 16) ? 8 : 0;
    const long rowBase = ((long)z * p.Ho + y) * p.Wo;
    const long nvox    = (long)p.Do * p.Ho * p.Wo;

    const float scale = p.gamma[co] * rsqrtf(p.var[co] + BN_EPS);
    const float shift = p.beta[co] - p.mean[co] * scale;
    const float bsv   = p.bias ? p.bias[co] : 0.f;
    float*      outp  = p.out + ((long)b * p.Cout + co) * nvox + rowBase;
    const float* mp   = p.mask ? (p.mask + (long)b * nvox + rowBase) : nullptr;

    #pragma unroll
    for (int r = 0; r < 8; ++r) {
        int x = x0 + mb + r;
        if (x < p.Wo) {
            float val = (acc0[r] + bsv) * scale + shift;
            if (p.relu) val = fmaxf(val, 0.f);
            if (mp) val *= mp[x];
            outp[x] = val;
        }
        int x2 = x + 16;
        if (x2 < p.Wo) {
            float val = (acc1[r] + bsv) * scale + shift;
            if (p.relu) val = fmaxf(val, 0.f);
            if (mp) val *= mp[x2];
            outp[x2] = val;
        }
    }
}

// ---------------------------------------------------------------------------
// Small elementwise / reduction kernels (division-free grid mappings)
// ---------------------------------------------------------------------------
__global__ void mask_from_occ_kernel(const int* occ, float* mask, long n) {
    long i = (long)blockIdx.x * blockDim.x + threadIdx.x;
    if (i < n) mask[i] = (occ[i] == 0) ? 1.f : 0.f;
}

// out[b,c,v] = feat[b,c,v] * mask[b,v]; grid = (vox_blocks, C, B)
__global__ void mul_mask_kernel(const float* feat, const float* mask, float* out,
                                int C, long nvox) {
    long v = (long)blockIdx.x * blockDim.x + threadIdx.x;
    if (v >= nvox) return;
    int  c = blockIdx.y, b = blockIdx.z;
    long i = ((long)b * C + c) * nvox + v;
    out[i] = feat[i] * mask[(long)b * nvox + v];
}

// mout[b,z,y,x] = any(mask over receptive field); grid = (Ho, Do, B), block=64
__global__ void mask_down_kernel(const float* mi, float* mo,
                                 int Di, int Hi, int Wi, int Do, int Ho, int Wo,
                                 int kd, int kh, int kw, int sd, int sh, int sw,
                                 int pd, int ph, int pw) {
    int x = threadIdx.x;
    if (x >= Wo) return;
    int y = blockIdx.x, z = blockIdx.y, b = blockIdx.z;
    float any = 0.f;
    for (int a = 0; a < kd; ++a) {
        int zi = z * sd - pd + a;
        if (zi < 0 || zi >= Di) continue;
        for (int c = 0; c < kh; ++c) {
            int yi = y * sh - ph + c;
            if (yi < 0 || yi >= Hi) continue;
            for (int d = 0; d < kw; ++d) {
                int xi = x * sw - pw + d;
                if (xi < 0 || xi >= Wi) continue;
                if (mi[((long)b * Di + zi) * Hi * Wi + (long)yi * Wi + xi] > 0.f)
                    any = 1.f;
            }
        }
    }
    mo[((long)b * Do + z) * Ho * Wo + (long)y * Wo + x] = any;
}

// s[b,c] = sum(x*mask) / max(sum(mask), 1); grid = (C, B)
__global__ void se_reduce_kernel(const float* x, const float* mask, float* s,
                                 int C, long nvox) {
    int c = blockIdx.x, b = blockIdx.y;
    const float* xp = x + ((long)b * C + c) * nvox;
    const float* mp = mask + (long)b * nvox;
    float sv = 0.f, mv = 0.f;
    for (long i = threadIdx.x; i < nvox; i += blockDim.x) {
        float m = mp[i];
        sv += xp[i] * m;
        mv += m;
    }
    __shared__ float s1[256], s2[256];
    s1[threadIdx.x] = sv;
    s2[threadIdx.x] = mv;
    __syncthreads();
    for (int o = 128; o > 0; o >>= 1) {
        if ((int)threadIdx.x < o) {
            s1[threadIdx.x] += s1[threadIdx.x + o];
            s2[threadIdx.x] += s2[threadIdx.x + o];
        }
        __syncthreads();
    }
    if (threadIdx.x == 0) s[(long)b * C + c] = s1[0] / fmaxf(s2[0], 1.f);
}

// y[b,c] = sigmoid( relu(s @ w1^T) @ w2^T ); grid = (B), block = 128
__global__ void se_fc_kernel(const float* s, const float* w1, const float* w2,
                             float* y, int C, int R) {
    int b = blockIdx.x;
    int t = threadIdx.x;
    __shared__ float hid[16];
    if (t < R) {
        float a = 0.f;
        for (int c = 0; c < C; ++c) a += s[(long)b * C + c] * w1[(long)t * C + c];
        hid[t] = fmaxf(a, 0.f);
    }
    __syncthreads();
    if (t < C) {
        float a = 0.f;
        for (int j = 0; j < R; ++j) a += hid[j] * w2[(long)t * R + j];
        y[(long)b * C + t] = 1.f / (1.f + __expf(-a));
    }
}

// out = relu(xse * y[b,c] + identity) * mask; grid = (vox_blocks, C, B)
__global__ void se_apply_kernel(const float* xse, const float* idn, const float* y,
                                const float* mask, float* out, int C, long nvox) {
    long v = (long)blockIdx.x * blockDim.x + threadIdx.x;
    if (v >= nvox) return;
    int  c = blockIdx.y, b = blockIdx.z;
    long i = ((long)b * C + c) * nvox + v;
    out[i] = fmaxf(xse[i] * y[(long)b * C + c] + idn[i], 0.f) *
             mask[(long)b * nvox + v];
}

// ---------------------------------------------------------------------------
// Host orchestration
// ---------------------------------------------------------------------------
struct BNP { const float *g, *b, *m, *v; };
struct Blk { const float *w1, *b1; BNP bn1; const float *w2, *b2; BNP bn2;
             const float *se1, *se2; };

extern "C" void kernel_launch(void* const* d_in, const int* in_sizes, int n_in,
                              void* d_out, int out_size, void* d_ws, size_t ws_size,
                              hipStream_t stream) {
    const int B = 2;
    const float* vfeat = (const float*)d_in[0];   // [2,5,31,100,100]
    const int*   occ   = (const int*)d_in[1];     // [2,31,100,100]

    // Walk params in setup_inputs() insertion order.
    int idx = 2;
    auto nxt = [&]() { return (const float*)d_in[idx++]; };
    auto getbn = [&]() { BNP bn; bn.g = nxt(); bn.b = nxt(); bn.m = nxt(); bn.v = nxt(); return bn; };
    auto getblk = [&]() {
        Blk k;
        k.w1 = nxt(); k.b1 = nxt(); k.bn1 = getbn();
        k.w2 = nxt(); k.b2 = nxt(); k.bn2 = getbn();
        k.se1 = nxt(); k.se2 = nxt();
        return k;
    };

    const float* ci_w; BNP ci_bn;
    ci_w = nxt(); ci_bn = getbn();
    Blk r1_0 = getblk(), r1_1 = getblk();
    const float* d2_w; BNP d2_bn; d2_w = nxt(); d2_bn = getbn();
    Blk r2_0 = getblk(), r2_1 = getblk();
    const float* d3_w; BNP d3_bn; d3_w = nxt(); d3_bn = getbn();
    Blk r3_0 = getblk(), r3_1 = getblk();
    const float* d4_w; BNP d4_bn; d4_w = nxt(); d4_bn = getbn();
    Blk r4_0 = getblk(), r4_1 = getblk();
    const float* co_w; BNP co_bn; co_w = nxt(); co_bn = getbn();

    // Workspace carve-out
    char* wsp = (char*)d_ws;
    auto alloc = [&](size_t bytes) {
        float* p = (float*)wsp;
        wsp += (bytes + 255) & ~(size_t)255;
        return p;
    };
    const long NV1 = 31L * 100 * 100;
    float* mask1 = alloc((size_t)B * NV1 * 4);
    float* mask2 = alloc((size_t)B * 16 * 50 * 50 * 4);
    float* mask3 = alloc((size_t)B * 8 * 25 * 25 * 4);
    float* mask4 = alloc((size_t)B * 3 * 13 * 13 * 4);
    float* mask5 = alloc((size_t)B * 1 * 13 * 13 * 4);
    float* sbuf  = alloc((size_t)B * 128 * 4);
    float* ybuf  = alloc((size_t)B * 128 * 4);
    size_t big   = (size_t)B * 16 * NV1 * 4;     // largest activation tensor
    float* cur = alloc(big);
    float* pB  = alloc(big);
    float* pC  = alloc(big);

    const float* curMask = mask1;
    int D = 31, H = 100, W = 100;

    auto conv = [&](const float* in, float* out, const float* w, const float* bias,
                    BNP bn, const float* mask, int Cin, int Cout,
                    int Di, int Hi, int Wi, int kd, int kh, int kw,
                    int sd, int sh, int sw, int pd, int ph, int pw, int relu) {
        int Do = (Di + 2 * pd - kd) / sd + 1;
        int Ho = (Hi + 2 * ph - kh) / sh + 1;
        int Wo = (Wi + 2 * pw - kw) / sw + 1;
        ConvParams p;
        p.in = in; p.out = out; p.w = w; p.bias = bias;
        p.gamma = bn.g; p.beta = bn.b; p.mean = bn.m; p.var = bn.v;
        p.mask = mask;
        p.Cin = Cin; p.Cout = Cout;
        p.Di = Di; p.Hi = Hi; p.Wi = Wi; p.Do = Do; p.Ho = Ho; p.Wo = Wo;
        p.sd = sd; p.sh = sh; p.sw = sw; p.pd = pd; p.ph = ph; p.pw = pw;
        p.relu = relu;
        p.nXTiles = (Wo + 31) / 32;
        dim3 grid((unsigned)(p.nXTiles * (Cout / 16)), (unsigned)(Do * Ho), B);
        if (kd == 3 && kh == 3 && kw == 3)
            conv3d_wmma_kernel<3, 3, 3><<<grid, dim3(32), 0, stream>>>(p);
        else
            conv3d_wmma_kernel<3, 1, 1><<<grid, dim3(32), 0, stream>>>(p);
    };

    auto resblock = [&](Blk& k, int C) {
        long nvox = (long)D * H * W;
        conv(cur, pB, k.w1, k.b1, k.bn1, curMask, C, C, D, H, W,
             3, 3, 3, 1, 1, 1, 1, 1, 1, /*relu=*/1);
        conv(pB, pC, k.w2, k.b2, k.bn2, curMask, C, C, D, H, W,
             3, 3, 3, 1, 1, 1, 1, 1, 1, /*relu=*/0);
        se_reduce_kernel<<<dim3(C, B), 256, 0, stream>>>(pC, curMask, sbuf, C, nvox);
        int R = C / 16; if (R < 1) R = 1;
        se_fc_kernel<<<B, 128, 0, stream>>>(sbuf, k.se1, k.se2, ybuf, C, R);
        dim3 g((unsigned)((nvox + 255) / 256), C, B);
        se_apply_kernel<<<g, 256, 0, stream>>>(pC, cur, ybuf, curMask, pB, C, nvox);
        float* t = cur; cur = pB; pB = t;  // result now in cur
    };

    auto down = [&](const float* w, BNP bn, int Cin, int Cout, float* newMask,
                    int kd, int kh, int kw, int sd, int sh, int sw,
                    int pd, int ph, int pw) {
        int Do = (D + 2 * pd - kd) / sd + 1;
        int Ho = (H + 2 * ph - kh) / sh + 1;
        int Wo = (W + 2 * pw - kw) / sw + 1;
        mask_down_kernel<<<dim3(Ho, Do, B), 64, 0, stream>>>(
            curMask, newMask, D, H, W, Do, Ho, Wo, kd, kh, kw, sd, sh, sw, pd, ph, pw);
        conv(cur, pB, w, nullptr, bn, newMask, Cin, Cout, D, H, W,
             kd, kh, kw, sd, sh, sw, pd, ph, pw, /*relu=*/1);
        curMask = newMask; D = Do; H = Ho; W = Wo;
        float* t = cur; cur = pB; pB = t;
    };

    // ---- pipeline ----
    long nMask = (long)B * NV1;
    mask_from_occ_kernel<<<(unsigned)((nMask + 255) / 256), 256, 0, stream>>>(
        occ, mask1, nMask);

    {
        dim3 g((unsigned)((NV1 + 255) / 256), 5, B);
        mul_mask_kernel<<<g, 256, 0, stream>>>(vfeat, mask1, pC, 5, NV1);
    }

    conv(pC, cur, ci_w, nullptr, ci_bn, mask1, 5, 16, D, H, W,
         3, 3, 3, 1, 1, 1, 1, 1, 1, /*relu=*/1);   // conv_input -> cur

    resblock(r1_0, 16);
    resblock(r1_1, 16);

    down(d2_w, d2_bn, 16, 32, mask2, 3, 3, 3, 2, 2, 2, 1, 1, 1);
    resblock(r2_0, 32);
    resblock(r2_1, 32);

    down(d3_w, d3_bn, 32, 64, mask3, 3, 3, 3, 2, 2, 2, 1, 1, 1);
    resblock(r3_0, 64);
    resblock(r3_1, 64);

    down(d4_w, d4_bn, 64, 128, mask4, 3, 3, 3, 2, 2, 2, 0, 1, 1);
    resblock(r4_0, 128);
    resblock(r4_1, 128);

    // conv_out: k(3,1,1) s(2,1,1) p0 -> [2,128,1,13,13] written to d_out
    {
        int kd = 3, kh = 1, kw = 1, sd = 2, sh = 1, sw = 1;
        int Do2 = (D - kd) / sd + 1, Ho2 = H, Wo2 = W;
        mask_down_kernel<<<dim3(Ho2, Do2, B), 64, 0, stream>>>(
            curMask, mask5, D, H, W, Do2, Ho2, Wo2, kd, kh, kw, sd, sh, sw, 0, 0, 0);
        conv(cur, (float*)d_out, co_w, nullptr, co_bn, mask5, 128, 128, D, H, W,
             kd, kh, kw, sd, sh, sw, 0, 0, 0, /*relu=*/1);
    }
    (void)in_sizes; (void)n_in; (void)out_size; (void)ws_size;
}
